// SupConLoss_438086664723
// MI455X (gfx1250) — matrix operand order repaired and checked by reading, not compiled, not run on hardware
//
#include <hip/hip_runtime.h>
#include <math.h>

typedef __attribute__((ext_vector_type(16))) _Float16 v16h;
typedef __attribute__((ext_vector_type(8)))  _Float16 v8h;
typedef __attribute__((ext_vector_type(8)))  float    v8f;

#define N_ROWS 16384
#define DIM    128
#define NCLS   10
#define TEMP   0.2f
// 5 * log2(e): exp(5*(c-1)) == exp2(L2E5*c - L2E5)
#define L2E5   7.2134752044448170367996234050095f

#define ROW_STRIDE 272                 // 128 f16 + 16B pad: bank-conflict-free
#define PANEL_BYTES (128 * ROW_STRIDE) // 34816 B per panel, x2 buffers

__device__ __forceinline__ float fast_exp2(float x) {
#if __has_builtin(__builtin_amdgcn_exp2f)
    return __builtin_amdgcn_exp2f(x);
#else
    return exp2f(x);
#endif
}

__device__ __forceinline__ void wait_async0() {
#if __has_builtin(__builtin_amdgcn_s_wait_asynccnt)
    __builtin_amdgcn_s_wait_asynccnt(0);
#else
    asm volatile("s_wait_asynccnt 0x0" ::: "memory");
#endif
}

// Copy 128 contiguous bytes global -> LDS with 8 async b128 transfers.
// IOFFSET is added to BOTH the LDS and global addresses (ISA 15.18.3).
__device__ __forceinline__ void async_copy_row128(unsigned lds, const void* g) {
    asm volatile(
        "global_load_async_to_lds_b128 %0, %1, off\n\t"
        "global_load_async_to_lds_b128 %0, %1, off offset:16\n\t"
        "global_load_async_to_lds_b128 %0, %1, off offset:32\n\t"
        "global_load_async_to_lds_b128 %0, %1, off offset:48\n\t"
        "global_load_async_to_lds_b128 %0, %1, off offset:64\n\t"
        "global_load_async_to_lds_b128 %0, %1, off offset:80\n\t"
        "global_load_async_to_lds_b128 %0, %1, off offset:96\n\t"
        "global_load_async_to_lds_b128 %0, %1, off offset:112"
        :: "v"(lds), "v"(g) : "memory");
}

// workspace layout
static constexpr size_t WS_FEATS = 0;                                          // _Float16[N][128]
static constexpr size_t WS_S     = (size_t)N_ROWS * DIM * sizeof(_Float16);    // float[16][128]
static constexpr size_t WS_CNT   = WS_S + 16 * DIM * sizeof(float);            // float[16]
static constexpr size_t WS_LOSS  = WS_CNT + 16 * sizeof(float);                // float[N]
static constexpr size_t WS_VALID = WS_LOSS + (size_t)N_ROWS * sizeof(float);   // float[N]

// ---------------------------------------------------------------------------
// Kernel 1: L2-normalize each row and convert to f16. One block per row.
// ---------------------------------------------------------------------------
__global__ __launch_bounds__(128) void k_normalize(const float* __restrict__ feats,
                                                   _Float16* __restrict__ out) {
    __shared__ float red[128];
    const int row = blockIdx.x;
    const int t   = threadIdx.x;
    float x = feats[(size_t)row * DIM + t];
    red[t] = x * x;
    __syncthreads();
    #pragma unroll
    for (int s = 64; s > 0; s >>= 1) {
        if (t < s) red[t] += red[t + s];
        __syncthreads();
    }
    float rn = 1.0f / sqrtf(red[0]);
    out[(size_t)row * DIM + t] = (_Float16)(x * rn);
}

// ---------------------------------------------------------------------------
// Kernel 2: per-class feature sums S_k = sum_{label==k} f_j and class counts.
// ---------------------------------------------------------------------------
__global__ __launch_bounds__(128) void k_classsum(const _Float16* __restrict__ feats,
                                                  const int* __restrict__ labels,
                                                  float* __restrict__ S,
                                                  float* __restrict__ cnt) {
    const int k = blockIdx.x;
    const int t = threadIdx.x;
    float acc = 0.f, c = 0.f;
    for (int i = 0; i < N_ROWS; ++i) {
        if (labels[i] == k) {
            acc += (float)feats[(size_t)i * DIM + t];
            c += 1.f;
        }
    }
    S[k * DIM + t] = acc;
    if (t == 0) cnt[k] = c;
}

// ---------------------------------------------------------------------------
// Kernel 3: main pass. 256 threads = 8 waves; each wave owns 32 rows (two
// 16-row M-tiles sharing every B fragment), block covers 256 rows. Panels are
// double-buffered in dynamic LDS and filled with async global->LDS DMA that
// overlaps the WMMA+exp compute; one s_wait_asynccnt + barrier per panel.
// ---------------------------------------------------------------------------
__global__ __launch_bounds__(256) void k_supcon(const _Float16* __restrict__ feats,
                                                const int*      __restrict__ labels,
                                                const float*    __restrict__ S,
                                                const float*    __restrict__ cnt,
                                                float* __restrict__ loss_row,
                                                float* __restrict__ valid_row) {
    extern __shared__ char smem[];   // 2 * PANEL_BYTES

    const int t    = threadIdx.x;
    const int lane = t & 31;
    const int wave = t >> 5;
    const int n16  = lane & 15;   // N index within 16x16 tile
    const int hi   = lane >> 4;   // lane-half selector

    const int i0   = blockIdx.x * 256 + wave * 32;   // this wave's first row
    const int rowA = i0 + n16;          // tile A rows: i0 .. i0+15
    const int rowB = i0 + 16 + n16;     // tile B rows: i0+16 .. i0+31

    // --- A fragments (16-bit 16x32 layout): lane m<16 holds K {kb..kb+7,
    //     kb+16..kb+23}; lane m+16 holds K {kb+8..kb+15, kb+24..kb+31}. ---
    v16h afA[4], afB[4];
    const _Float16* baA = feats + (size_t)rowA * DIM;
    const _Float16* baB = feats + (size_t)rowB * DIM;
    #pragma unroll
    for (int ks = 0; ks < 4; ++ks) {
        const int kb = ks * 32;
        v8h lo = *(const v8h*)(baA + kb + 8 * hi);
        v8h hp = *(const v8h*)(baA + kb + 16 + 8 * hi);
        afA[ks] = __builtin_shufflevector(lo, hp,
            0, 1, 2, 3, 4, 5, 6, 7, 8, 9, 10, 11, 12, 13, 14, 15);
        lo = *(const v8h*)(baB + kb + 8 * hi);
        hp = *(const v8h*)(baB + kb + 16 + 8 * hi);
        afB[ks] = __builtin_shufflevector(lo, hp,
            0, 1, 2, 3, 4, 5, 6, 7, 8, 9, 10, 11, 12, 13, 14, 15);
    }

    // Self-dots and class-sum dots (lane pair L/L+16 holds all 128 K values).
    const float* SA = S + labels[rowA] * DIM;
    const float* SB = S + labels[rowB] * DIM;
    float sdA = 0.f, sdB = 0.f, dcA = 0.f, dcB = 0.f;
    #pragma unroll
    for (int ks = 0; ks < 4; ++ks) {
        const int kb = ks * 32;
        #pragma unroll
        for (int q = 0; q < 8; ++q) {
            const int k0 = kb + 8 * hi + q;        // frag elems 0..7
            const int k1 = kb + 16 + 8 * hi + q;   // frag elems 8..15
            float a0 = (float)afA[ks][q],  a1 = (float)afA[ks][q + 8];
            float b0 = (float)afB[ks][q],  b1 = (float)afB[ks][q + 8];
            sdA = fmaf(a0, a0, sdA);  sdA = fmaf(a1, a1, sdA);
            sdB = fmaf(b0, b0, sdB);  sdB = fmaf(b1, b1, sdB);
            dcA = fmaf(a0, SA[k0], dcA);  dcA = fmaf(a1, SA[k1], dcA);
            dcB = fmaf(b0, SB[k0], dcB);  dcB = fmaf(b1, SB[k1], dcB);
        }
    }
    sdA += __shfl_xor(sdA, 16, 32);  sdB += __shfl_xor(sdB, 16, 32);
    dcA += __shfl_xor(dcA, 16, 32);  dcB += __shfl_xor(dcB, 16, 32);

    float seA[8], seB[8];
    #pragma unroll
    for (int r = 0; r < 8; ++r) { seA[r] = 0.f; seB[r] = 0.f; }

    // -- async double-buffered panel staging --
    const int jr = t >> 1, seg = t & 1;   // each thread owns half of one row
    const unsigned lds_my =
        (unsigned)(uintptr_t)smem + (unsigned)(jr * ROW_STRIDE + seg * 128);

    // prefetch panel 0 into buffer 0
    async_copy_row128(lds_my, feats + (size_t)jr * DIM + seg * 64);

    const int NPANEL = N_ROWS / 128;
    for (int p = 0; p < NPANEL; ++p) {
        wait_async0();      // this wave's DMA into buf[p&1] complete
        __syncthreads();    // all waves' DMA done; all done reading buf[(p+1)&1]'s
                            // previous contents
        if (p + 1 < NPANEL) {
            async_copy_row128(lds_my + ((p + 1) & 1) * PANEL_BYTES,
                feats + (size_t)((p + 1) * 128 + jr) * DIM + seg * 64);
        }
        const char* pbase = smem + (p & 1) * PANEL_BYTES;

        #pragma unroll 1
        for (int jt = 0; jt < 8; ++jt) {
            const int jloc = jt * 16 + n16;
            const _Float16* brow = (const _Float16*)(pbase + jloc * ROW_STRIDE);
            // B layout (16-bit 32x16): lane n<16 holds K {0..7,16..23}, lane n+16 K {8..15,24..31}
            v8f cA = {}, cB = {};
            #pragma unroll
            for (int ks = 0; ks < 4; ++ks) {
                v8h lo = *(const v8h*)(brow + ks * 32 + 8 * hi);
                v8h hp = *(const v8h*)(brow + ks * 32 + 16 + 8 * hi);
                v16h b = __builtin_shufflevector(lo, hp,
                    0, 1, 2, 3, 4, 5, 6, 7, 8, 9, 10, 11, 12, 13, 14, 15);
                cA = __builtin_amdgcn_wmma_f32_16x16x32_f16(
                         false, afA[ks], false, b, (short)0, cA, false, false);
                cB = __builtin_amdgcn_wmma_f32_16x16x32_f16(
                         false, afB[ks], false, b, (short)0, cB, false, false);
            }
            // Denominator only: exp(5*(c-1)) accumulated per row.
            #pragma unroll
            for (int r = 0; r < 8; ++r)
                seA[r] += fast_exp2(fmaf(cA[r], L2E5, -L2E5));
            #pragma unroll
            for (int r = 0; r < 8; ++r)
                seB[r] += fast_exp2(fmaf(cB[r], L2E5, -L2E5));
        }
    }

    // reduce across the 16-lane N dimension (xor masks stay within each half)
    #pragma unroll
    for (int r = 0; r < 8; ++r) {
        #pragma unroll
        for (int m = 8; m >= 1; m >>= 1) {
            seA[r] += __shfl_xor(seA[r], m, 32);
            seB[r] += __shfl_xor(seB[r], m, 32);
        }
    }

    // Broadcast per-row scalars (full EXEC required: before divergent branch).
    float sdrA[8], sdrB[8], dcrA[8], dcrB[8];
    #pragma unroll
    for (int r = 0; r < 8; ++r) {
        sdrA[r] = __shfl(sdA, 8 * hi + r, 32);
        sdrB[r] = __shfl(sdB, 8 * hi + r, 32);
        dcrA[r] = __shfl(dcA, 8 * hi + r, 32);
        dcrB[r] = __shfl(dcB, 8 * hi + r, 32);
    }

    if (n16 == 0) {   // lanes 0 and 16 hold rows {i0, i0+8} + r etc.
        #pragma unroll
        for (int r = 0; r < 8; ++r) {
            {
                const int row   = i0 + 8 * hi + r;
                const float sdv = sdrA[r];
                const float np  = cnt[labels[row]] - 1.f;
                const float sexp = seA[r] - fast_exp2(fmaf(sdv, L2E5, -L2E5));
                const float spos = 5.f * (dcrA[r] - sdv - np);
                const float loss = (np > 0.f)
                    ? TEMP * (np * logf(sexp) - spos) / np : 0.f;
                loss_row[row]  = loss;
                valid_row[row] = (np > 0.f) ? 1.f : 0.f;
            }
            {
                const int row   = i0 + 16 + 8 * hi + r;
                const float sdv = sdrB[r];
                const float np  = cnt[labels[row]] - 1.f;
                const float sexp = seB[r] - fast_exp2(fmaf(sdv, L2E5, -L2E5));
                const float spos = 5.f * (dcrB[r] - sdv - np);
                const float loss = (np > 0.f)
                    ? TEMP * (np * logf(sexp) - spos) / np : 0.f;
                loss_row[row]  = loss;
                valid_row[row] = (np > 0.f) ? 1.f : 0.f;
            }
        }
    }
}

// ---------------------------------------------------------------------------
// Kernel 4: deterministic final reduction to scalar loss.
// ---------------------------------------------------------------------------
__global__ __launch_bounds__(256) void k_reduce(const float* __restrict__ loss_row,
                                                const float* __restrict__ valid_row,
                                                float* __restrict__ out) {
    __shared__ float sl[256], sv[256];
    const int t = threadIdx.x;
    float a = 0.f, b = 0.f;
    for (int i = t; i < N_ROWS; i += 256) { a += loss_row[i]; b += valid_row[i]; }
    sl[t] = a; sv[t] = b;
    __syncthreads();
    #pragma unroll
    for (int s = 128; s > 0; s >>= 1) {
        if (t < s) { sl[t] += sl[t + s]; sv[t] += sv[t + s]; }
        __syncthreads();
    }
    if (t == 0) out[0] = sl[0] / fmaxf(sv[0], 1.0f);
}

extern "C" void kernel_launch(void* const* d_in, const int* in_sizes, int n_in,
                              void* d_out, int out_size, void* d_ws, size_t ws_size,
                              hipStream_t stream) {
    (void)in_sizes; (void)n_in; (void)out_size; (void)ws_size;
    const float* feats  = (const float*)d_in[0];
    const int*   labels = (const int*)d_in[1];

    _Float16* featsH    = (_Float16*)((char*)d_ws + WS_FEATS);
    float*    Ssum      = (float*)((char*)d_ws + WS_S);
    float*    cnt       = (float*)((char*)d_ws + WS_CNT);
    float*    loss_row  = (float*)((char*)d_ws + WS_LOSS);
    float*    valid_row = (float*)((char*)d_ws + WS_VALID);

    k_normalize<<<N_ROWS, 128, 0, stream>>>(feats, featsH);
    k_classsum<<<NCLS, 128, 0, stream>>>(featsH, labels, Ssum, cnt);
    k_supcon<<<N_ROWS / 256, 256, 2 * PANEL_BYTES, stream>>>(
        featsH, labels, Ssum, cnt, loss_row, valid_row);
    k_reduce<<<1, 256, 0, stream>>>(loss_row, valid_row, (float*)d_out);
}
